// EdgePredictorGNN_30090540876016
// MI455X (gfx1250) — matrix operand order repaired and checked by reading, not compiled, not run on hardware
//
#include <hip/hip_runtime.h>
#include <hip/hip_bf16.h>

#define NN 100000
#define NE 1600000
#define CH 128

typedef __attribute__((ext_vector_type(16))) __bf16 v16bf;
typedef __attribute__((ext_vector_type(8)))  __bf16 v8bf;
typedef __attribute__((ext_vector_type(8)))  float  v8f;
typedef __attribute__((ext_vector_type(4)))  unsigned v4u;
typedef __attribute__((ext_vector_type(8)))  int    v8i;
typedef __attribute__((ext_vector_type(4)))  int    v4i;

__device__ __forceinline__ v8f wmma_bf16(v16bf a, v16bf b, v8f c) {
    return __builtin_amdgcn_wmma_f32_16x16x32_bf16(
        /*neg_a=*/false, a, /*neg_b=*/false, b,
        /*c_mod=*/(short)0, c, /*reuse_a=*/false, /*reuse_b=*/false);
}

// TDM: DMA one contiguous 64KB block (16384 dwords) from global to LDS offset
// lds_off. D# packed per CDNA5 ISA:
//  group0: count=1, lds_addr, global_addr[56:0], type=2
//  group1: data_size=4B, tensor_dim0=16384, tensor_dim1=1, tile_dim0=16384,
//          tile_dim1=1, tensor_dim0_stride=16384
__device__ __forceinline__ void tdm_load_64k(const void* gsrc, unsigned lds_off) {
    unsigned long long ga = (unsigned long long)(uintptr_t)gsrc;
    v4u g0;
    g0[0] = 1u;                                            // count=1 (user D#)
    g0[1] = lds_off;                                       // lds_addr (bytes)
    g0[2] = (unsigned)(ga & 0xffffffffu);                  // global_addr lo
    g0[3] = (unsigned)((ga >> 32) & 0x01ffffffu) | (2u << 30); // addr hi | type=2
    v8i g1;
    g1[0] = (2 << 16);                 // workgroup_mask=0, data_size=2 (4 bytes)
    g1[1] = (int)(16384u << 16);       // tensor_dim0[15:0]=16384 in bits [63:48]
    g1[2] = (1 << 16);                 // tensor_dim0 hi=0, tensor_dim1[15:0]=1
    g1[3] = (int)(16384u << 16);       // tensor_dim1 hi=0, tile_dim0=16384
    g1[4] = 1;                         // tile_dim1=1, tile_dim2=0
    g1[5] = 16384;                     // tensor_dim0_stride lo32
    g1[6] = 0;                         // stride hi, tensor_dim1_stride lo
    g1[7] = 0;
    v4i g2 = {0, 0, 0, 0};
    v4i g3 = {0, 0, 0, 0};
#if defined(__clang_major__) && (__clang_major__ >= 23)
    v8i gz = {0, 0, 0, 0, 0, 0, 0, 0};
    __builtin_amdgcn_tensor_load_to_lds(g0, g1, g2, g3, gz, 0);
#else
    __builtin_amdgcn_tensor_load_to_lds(g0, g1, g2, g3, 0);
#endif
}

// A-operand (16x32 bf16) per-lane loader from an f32 row with scaling.
__device__ __forceinline__ v16bf load_a_f32(const float* p, float s) {
    float4 a = *(const float4*)(p);
    float4 b = *(const float4*)(p + 4);
    float4 c = *(const float4*)(p + 16);
    float4 d = *(const float4*)(p + 20);
    float f[16] = {a.x,a.y,a.z,a.w, b.x,b.y,b.z,b.w,
                   c.x,c.y,c.z,c.w, d.x,d.y,d.z,d.w};
    v16bf r;
#pragma unroll
    for (int j = 0; j < 16; ++j) r[j] = (__bf16)(f[j] * s);
    return r;
}

__device__ __forceinline__ v16bf load_a_bf16(const __bf16* p) {
    v8bf lo = *(const v8bf*)(p);
    v8bf hi = *(const v8bf*)(p + 16);
    return __builtin_shufflevector(lo, hi, 0,1,2,3,4,5,6,7,8,9,10,11,12,13,14,15);
}

__global__ void zero_f32(float* __restrict__ p, int n) {
    int i = blockIdx.x * blockDim.x + threadIdx.x;
    int stride = gridDim.x * blockDim.x;
    for (; i < n; i += stride) p[i] = 0.0f;
}

// Swizzle 128x128 f32 weights (row-major [K][N]) into per-lane WMMA B-operand
// layout: sw[w][kt][nt][lane][16], 16 contiguous bf16 per lane.
// B layout (16-bit, 32x16): lanes 0-15 col N=lane, K=kt*32+{0..7,16..23};
// lanes 16-31 col N=lane-16, K=kt*32+{8..15,24..31}.
__global__ void prep_weights(const float* __restrict__ Wl0, const float* __restrict__ Wr0,
                             const float* __restrict__ Wl1, const float* __restrict__ Wr1,
                             const float* __restrict__ Wd1, __bf16* __restrict__ sw) {
    int t = blockIdx.x * 256 + threadIdx.x;
    if (t >= 6 * 4 * 8 * 32) return;
    int lane = t & 31;
    int nt   = (t >> 5) & 7;
    int kt   = (t >> 8) & 3;
    int w    = t >> 10;
    const float* W;
    if      (w == 0) W = Wl0;
    else if (w == 1) W = Wr0;
    else if (w == 2) W = Wl1;
    else if (w == 3) W = Wr1;
    else if (w == 4) W = Wd1;            // Wd1 rows 0..127   (src half)
    else             W = Wd1 + 128 * CH; // Wd1 rows 128..255 (dst half)
    int n  = nt * 16 + (lane & 15);
    int kb = kt * 32 + ((lane & 16) ? 8 : 0);
    __bf16* d = sw + (size_t)w * 16384 + (size_t)(((kt * 8 + nt) * 32 + lane) * 16);
#pragma unroll
    for (int j = 0; j < 16; ++j) d[j] = (__bf16)W[(kb + j + (j >= 8 ? 8 : 0)) * CH + n];
}

__global__ void count_kernel(const int* __restrict__ dst, float* __restrict__ cnt, int nE) {
    int e = blockIdx.x * 256 + threadIdx.x;
    if (e < nE) atomicAdd(cnt + dst[e], 1.0f);
}

__global__ void scatter_kernel(const float* __restrict__ feat, const int* __restrict__ src,
                               const int* __restrict__ dst, float* __restrict__ agg, int nE) {
    int tid = blockIdx.x * 256 + threadIdx.x;
    int e = tid >> 5;
    if (e >= nE) return;
    int c = (tid & 31) << 2;
    int s = src[e], d = dst[e];
    float4 v = *(const float4*)(feat + (size_t)s * CH + c);
    float* o = agg + (size_t)d * CH + c;
    atomicAdd(o + 0, v.x);
    atomicAdd(o + 1, v.y);
    atomicAdd(o + 2, v.z);
    atomicAdd(o + 3, v.w);
}

// Fused SAGEConv GEMM: out = act(mean(agg) @ Wl + bias + feat @ Wr)
// 256 threads = 8 wave-tiles of 16 rows; weights (Wl|Wr contiguous, 64KB)
// TDM-staged into LDS once per block, B operands served from ds_load.
template <bool RELU, bool OUTBF16>
__global__ __launch_bounds__(256)
void sage_gemm(const float* __restrict__ feat, const float* __restrict__ agg,
               const float* __restrict__ cnt, const __bf16* __restrict__ Wlr,
               const float* __restrict__ bias,
               float* __restrict__ outF, __bf16* __restrict__ outB) {
    __shared__ __bf16 lw[32768];   // 64KB: Wl then Wr
    if (threadIdx.x < 32) {
        tdm_load_64k(Wlr, 0);
        __builtin_amdgcn_s_wait_tensorcnt(0);
    }
    __syncthreads();
    // TDM writes LDS behind the compiler's back: escape the array address and
    // declare a memory clobber so ds_loads below can't be folded to undef.
    asm volatile("" :: "s"(lw) : "memory");
    const __bf16* Wl = lw;
    const __bf16* Wr = lw + 16384;

    const int lane  = threadIdx.x & 31;
    const int rbase = blockIdx.x * 128 + (threadIdx.x >> 5) * 16;
    if (rbase >= NN) return;
    const int row   = rbase + (lane & 15);
    const float scale = 1.0f / fmaxf(cnt[row], 1.0f);
    const int kofs  = (lane & 16) ? 8 : 0;
    const float* arow = agg  + (size_t)row * CH;
    const float* xrow = feat + (size_t)row * CH;
    v8f acc[8] = {};
#pragma unroll
    for (int kt = 0; kt < 4; ++kt) {
        const int kb = kt * 32 + kofs;
        v16bf aA = load_a_f32(arow + kb, scale);
        v16bf aX = load_a_f32(xrow + kb, 1.0f);
#pragma unroll
        for (int nt = 0; nt < 8; ++nt) {
            const int bo = ((kt * 8 + nt) * 32 + lane) * 16;
            v16bf bl = *(const v16bf*)(Wl + bo);
            v16bf br = *(const v16bf*)(Wr + bo);
            acc[nt] = wmma_bf16(aA, bl, acc[nt]);
            acc[nt] = wmma_bf16(aX, br, acc[nt]);
        }
    }
    // C/D layout: lanes 0-15: N=lane, M=v; lanes 16-31: N=lane-16, M=v+8
    const int col0 = lane & 15;
    const int rofs = (lane & 16) ? 8 : 0;
#pragma unroll
    for (int nt = 0; nt < 8; ++nt) {
        const int col = nt * 16 + col0;
        const float b = bias[col];
#pragma unroll
        for (int v = 0; v < 8; ++v) {
            float r = acc[nt][v] + b;
            if (RELU) r = fmaxf(r, 0.0f);
            const size_t o = (size_t)(rbase + rofs + v) * CH + col;
            if (OUTBF16) outB[o] = (__bf16)r;
            else         outF[o] = r;
        }
    }
}

// Decoder: 256 threads = 8 wave-tiles of 16 edges. Wd1 halves (64KB) staged
// via TDM into LDS. h1 = relu(z[src]@Ws + z[dst]@Wd + bd1); logit = h1.wd2+bd2.
__global__ __launch_bounds__(256)
void decoder_kernel(const __bf16* __restrict__ z, const int* __restrict__ src,
                    const int* __restrict__ dst, const __bf16* __restrict__ Wsd,
                    const float* __restrict__ bd1, const float* __restrict__ wd2,
                    const float* __restrict__ bd2, float* __restrict__ out) {
    __shared__ __bf16 lw[32768];   // 64KB: Ws then Wd
    if (threadIdx.x < 32) {
        tdm_load_64k(Wsd, 0);
        __builtin_amdgcn_s_wait_tensorcnt(0);
    }
    __syncthreads();
    // TDM writes LDS behind the compiler's back: escape + memory clobber so
    // the B-tile ds_loads are treated as reading real (externally written) data.
    asm volatile("" :: "s"(lw) : "memory");
    const __bf16* Ws = lw;
    const __bf16* Wd = lw + 16384;

    const int lane  = threadIdx.x & 31;
    const int ebase = blockIdx.x * 128 + (threadIdx.x >> 5) * 16;
    const int e     = ebase + (lane & 15);
    const __bf16* zs = z + (size_t)src[e] * CH;
    const __bf16* zd = z + (size_t)dst[e] * CH;
    const int kofs = (lane & 16) ? 8 : 0;
    v8f acc[8] = {};
#pragma unroll
    for (int kt = 0; kt < 4; ++kt) {
        const int kb = kt * 32 + kofs;
        v16bf as = load_a_bf16(zs + kb);
        v16bf ad = load_a_bf16(zd + kb);
#pragma unroll
        for (int nt = 0; nt < 8; ++nt) {
            const int bo = ((kt * 8 + nt) * 32 + lane) * 16;
            v16bf bs = *(const v16bf*)(Ws + bo);
            v16bf bd = *(const v16bf*)(Wd + bo);
            acc[nt] = wmma_bf16(as, bs, acc[nt]);
            acc[nt] = wmma_bf16(ad, bd, acc[nt]);
        }
    }
    const int col0 = lane & 15;
    float p[8] = {};
#pragma unroll
    for (int nt = 0; nt < 8; ++nt) {
        const int col = nt * 16 + col0;
        const float b = bd1[col];
        const float w = wd2[col];
#pragma unroll
        for (int v = 0; v < 8; ++v)
            p[v] += fmaxf(acc[nt][v] + b, 0.0f) * w;
    }
    // reduce across each 16-lane half (xor masks < 16 stay within halves)
#pragma unroll
    for (int m = 1; m <= 8; m <<= 1) {
#pragma unroll
        for (int v = 0; v < 8; ++v)
            p[v] += __shfl_xor(p[v], m, 32);
    }
    const float b2 = bd2[0];
    if (lane == 0) {
#pragma unroll
        for (int v = 0; v < 8; ++v) out[ebase + v] = p[v] + b2;
    } else if (lane == 16) {
#pragma unroll
        for (int v = 0; v < 8; ++v) out[ebase + 8 + v] = p[v] + b2;
    }
}

extern "C" void kernel_launch(void* const* d_in, const int* in_sizes, int n_in,
                              void* d_out, int out_size, void* d_ws, size_t ws_size,
                              hipStream_t stream) {
    (void)in_sizes; (void)n_in; (void)out_size; (void)ws_size;
    const float* x    = (const float*)d_in[0];
    const int*   ei   = (const int*)d_in[1];
    const float* Wl0  = (const float*)d_in[2];
    const float* bl0  = (const float*)d_in[3];
    const float* Wr0  = (const float*)d_in[4];
    const float* Wl1  = (const float*)d_in[5];
    const float* bl1  = (const float*)d_in[6];
    const float* Wr1  = (const float*)d_in[7];
    const float* Wd1  = (const float*)d_in[8];
    const float* bd1  = (const float*)d_in[9];
    const float* Wd2  = (const float*)d_in[10];
    const float* bd2  = (const float*)d_in[11];
    const int* srcp = ei;
    const int* dstp = ei + NE;

    // workspace layout (agg and cnt contiguous so one zero pass covers both)
    char* ws = (char*)d_ws;
    const size_t AGG_B = (size_t)NN * CH * sizeof(float);   // 51.2 MB
    const size_t CNT_B = (size_t)NN * sizeof(float);        // 0.4 MB
    const size_t H_B   = (size_t)NN * CH * sizeof(float);   // 51.2 MB
    const size_t Z_B   = (size_t)NN * CH * sizeof(__bf16);  // 25.6 MB
    float*  agg = (float*)(ws);
    float*  cnt = (float*)(ws + AGG_B);
    float*  h   = (float*)(ws + AGG_B + CNT_B);
    __bf16* zb  = (__bf16*)(ws + AGG_B + CNT_B + H_B);
    __bf16* sw  = (__bf16*)(ws + AGG_B + CNT_B + H_B + Z_B);
    // contiguous 64KB pairs by construction (each block = 16384 bf16 = 32KB):
    __bf16* swL0 = sw;               // Wl0 | Wr0
    __bf16* swL1 = sw + 2 * 16384;   // Wl1 | Wr1
    __bf16* swD  = sw + 4 * 16384;   // Wd1 src-half | Wd1 dst-half

    // zero agg + cnt (contiguous), swizzle weights, build degree counts
    zero_f32<<<2048, 256, 0, stream>>>(agg, NN * CH + NN);
    prep_weights<<<24, 256, 0, stream>>>(Wl0, Wr0, Wl1, Wr1, Wd1, sw);
    count_kernel<<<(NE + 255) / 256, 256, 0, stream>>>(dstp, cnt, NE);

    // layer 0: scatter-add x, then fused mean/GEMM/ReLU -> h (f32)
    scatter_kernel<<<(NE * 32) / 256, 256, 0, stream>>>(x, srcp, dstp, agg, NE);
    sage_gemm<true, false><<<(NN + 127) / 128, 256, 0, stream>>>(x, agg, cnt, swL0, bl0, h, nullptr);

    // layer 1: re-zero agg, scatter-add h, fused GEMM -> z (bf16 for decoder)
    zero_f32<<<2048, 256, 0, stream>>>(agg, NN * CH);
    scatter_kernel<<<(NE * 32) / 256, 256, 0, stream>>>(h, srcp, dstp, agg, NE);
    sage_gemm<false, true><<<(NN + 127) / 128, 256, 0, stream>>>(h, agg, cnt, swL1, bl1, nullptr, zb);

    // decoder: per-16-edge WMMA tile + fused MLP head
    decoder_kernel<<<NE / 128, 256, 0, stream>>>(zb, srcp, dstp, swD, bd1, Wd2, bd2,
                                                 (float*)d_out);
}
